// MultiheadAttention_19834158973230
// MI455X (gfx1250) — compile-verified, matrix operand order checked by reading, not run on hardware
//
#include <hip/hip_runtime.h>

// ---------------------------------------------------------------------------
// MHA forward for T=512, B=16, E=768, H=12, D=64 on gfx1250 (wave32, WMMA).
// Pipeline: 3x WMMA GEMM (QKV proj, f32->f16) -> flash-attention WMMA kernel
// (bias streamed once, online softmax, K/V tiles staged to LDS via async
// copy engine) -> WMMA GEMM out-projection (f32 out).
// ---------------------------------------------------------------------------

typedef __attribute__((ext_vector_type(16))) _Float16 v16h;
typedef __attribute__((ext_vector_type(8)))  _Float16 v8h;
typedef __attribute__((ext_vector_type(2)))  _Float16 h2;
typedef __attribute__((ext_vector_type(8)))  float    v8f;
typedef __attribute__((ext_vector_type(4)))  int      v4i;

constexpr int Tn = 512;
constexpr int Bn = 16;
constexpr int En = 768;
constexpr int Hn = 12;
constexpr int Dn = 64;
constexpr int Mn = Tn * Bn;   // 8192 rows for the projections

// ----- gfx1250 async global->LDS copy (ASYNCcnt), guarded by __has_builtin --
#if __has_builtin(__builtin_amdgcn_global_load_async_to_lds_b128)
#define HAVE_ASYNC_LDS 1
#else
#define HAVE_ASYNC_LDS 0
#endif

// Builtin prototype (from the round-2 diagnostic):
//   (v4i addrspace(1)* gsrc, v4i addrspace(3)* ldst, imm offset, imm cpol)
typedef __attribute__((address_space(1))) v4i gas_v4i;
typedef __attribute__((address_space(3))) v4i las_v4i;

// Copy 16 bytes (8 halves) per lane from global to LDS.
__device__ __forceinline__ void async_copy16(const _Float16* g, _Float16* l) {
#if HAVE_ASYNC_LDS
  __builtin_amdgcn_global_load_async_to_lds_b128((gas_v4i*)g, (las_v4i*)l, 0, 0);
#else
  *(v8h*)l = *(const v8h*)g;
#endif
}

__device__ __forceinline__ void wait_async() {
#if HAVE_ASYNC_LDS
#if __has_builtin(__builtin_amdgcn_s_wait_asynccnt)
  __builtin_amdgcn_s_wait_asynccnt(0);
#else
  asm volatile("s_wait_asynccnt 0" ::: "memory");
#endif
#endif
}

// Load a 16-bit WMMA A/B fragment for one 16x32 (or 32x16) tile.
// Per-lane layout (CDNA5 ISA 7.12.2): elements 0..7 hold K = hi*8 + 0..7,
// elements 8..15 hold K = 16 + hi*8 + 0..7, where hi = lane>>4.
// `base` points at this lane's row/col start for the 32-wide K chunk.
__device__ __forceinline__ v16h load_frag16(const _Float16* base, int lhi) {
  v8h lo = *(const v8h*)(base + lhi * 8);
  v8h hi = *(const v8h*)(base + 16 + lhi * 8);
  v16h f;
#pragma unroll
  for (int i = 0; i < 8; ++i) { f[i] = lo[i]; f[i + 8] = hi[i]; }
  return f;
}

// ---------------------------------------------------------------------------
// Generic tiled GEMM:  C[m,n] = (sum_k A[m,k] * W[n,k] + bias[n]) * scale
// Block = 256 threads (8 waves), tile 128x128, K-step 32.
// Wave grid 2(M) x 4(N); each wave: 4x2 accumulators of 16x16.
// OUT_MODE 0: f32 row-major [M,N]
// OUT_MODE 1: f16 [B,H,T,D]  (m = t*B + b, n = h*64 + d)
// OUT_MODE 2: f16 [B,H,D,T]
// ---------------------------------------------------------------------------
template <bool AF16, int OUT_MODE>
__global__ __launch_bounds__(256) void gemm_wmma_kernel(
    const void* __restrict__ Ap, const float* __restrict__ W,
    const float* __restrict__ bias, void* __restrict__ Cp,
    int M, int N, int K, float scale) {
  constexpr int LDA = 40;  // 32 k-columns + 8 pad halves (80B row stride)
  __shared__ _Float16 lA[128 * LDA];
  __shared__ _Float16 lB[128 * LDA];

  const int tid  = threadIdx.x;
  const int wave = tid >> 5;
  const int lane = tid & 31;
  const int lrow = lane & 15;
  const int lhi  = lane >> 4;
  const int wm   = wave >> 2;  // 0..1
  const int wn   = wave & 3;   // 0..3
  const int bm   = blockIdx.x * 128;
  const int bn   = blockIdx.y * 128;
  const int lr   = tid >> 1;        // cooperative-load row 0..127
  const int lc   = (tid & 1) * 16;  // which 16-wide half of the k-step

  v8f zero = {};
  v8f acc[4][2];
#pragma unroll
  for (int i = 0; i < 4; ++i)
#pragma unroll
    for (int j = 0; j < 2; ++j) acc[i][j] = zero;

  for (int k0 = 0; k0 < K; k0 += 32) {
    // ---- stage A tile (128 x 32) into LDS as f16 ----
    if (AF16) {
      // A already f16: stage with the async copy engine (no conversion).
      const _Float16* As = (const _Float16*)Ap + (size_t)(bm + lr) * K + k0 + lc;
      async_copy16(As,     &lA[lr * LDA + lc]);
      async_copy16(As + 8, &lA[lr * LDA + lc + 8]);
    } else {
      const float* As = (const float*)Ap + (size_t)(bm + lr) * K + k0 + lc;
      float4 f0 = *(const float4*)(As + 0);
      float4 f1 = *(const float4*)(As + 4);
      float4 f2 = *(const float4*)(As + 8);
      float4 f3 = *(const float4*)(As + 12);
      float fv[16] = {f0.x, f0.y, f0.z, f0.w, f1.x, f1.y, f1.z, f1.w,
                      f2.x, f2.y, f2.z, f2.w, f3.x, f3.y, f3.z, f3.w};
      h2* dst = (h2*)&lA[lr * LDA + lc];
#pragma unroll
      for (int i = 0; i < 8; ++i) {
        h2 t; t[0] = (_Float16)fv[2 * i]; t[1] = (_Float16)fv[2 * i + 1]; dst[i] = t;
      }
    }
    // ---- stage W tile (128 x 32) into LDS as f16 (f32 -> f16 convert) ----
    {
      const float* Ws = W + (size_t)(bn + lr) * K + k0 + lc;
      float4 g0 = *(const float4*)(Ws + 0);
      float4 g1 = *(const float4*)(Ws + 4);
      float4 g2 = *(const float4*)(Ws + 8);
      float4 g3 = *(const float4*)(Ws + 12);
      float gv[16] = {g0.x, g0.y, g0.z, g0.w, g1.x, g1.y, g1.z, g1.w,
                      g2.x, g2.y, g2.z, g2.w, g3.x, g3.y, g3.z, g3.w};
      h2* dst = (h2*)&lB[lr * LDA + lc];
#pragma unroll
      for (int i = 0; i < 8; ++i) {
        h2 t; t[0] = (_Float16)gv[2 * i]; t[1] = (_Float16)gv[2 * i + 1]; dst[i] = t;
      }
    }
    if (AF16) wait_async();
    __syncthreads();

    // software prefetch of the next K-step (global_prefetch_b8)
    if (k0 + 32 < K) {
      if (AF16)
        __builtin_prefetch((const _Float16*)Ap + (size_t)(bm + lr) * K + k0 + 32 + lc, 0, 1);
      else
        __builtin_prefetch((const float*)Ap + (size_t)(bm + lr) * K + k0 + 32 + lc, 0, 1);
      __builtin_prefetch(W + (size_t)(bn + lr) * K + k0 + 32 + lc, 0, 1);
    }

    v16h af[4], bf[2];
#pragma unroll
    for (int mi = 0; mi < 4; ++mi)
      af[mi] = load_frag16(&lA[(wm * 64 + mi * 16 + lrow) * LDA], lhi);
#pragma unroll
    for (int ni = 0; ni < 2; ++ni)
      bf[ni] = load_frag16(&lB[(wn * 32 + ni * 16 + lrow) * LDA], lhi);
#pragma unroll
    for (int mi = 0; mi < 4; ++mi)
#pragma unroll
      for (int ni = 0; ni < 2; ++ni)
        acc[mi][ni] = __builtin_amdgcn_wmma_f32_16x16x32_f16(
            false, af[mi], false, bf[ni], (short)0, acc[mi][ni], false, false);
    __syncthreads();
  }

  // ---- epilogue: bias, scale, store in requested layout ----
#pragma unroll
  for (int mi = 0; mi < 4; ++mi) {
#pragma unroll
    for (int ni = 0; ni < 2; ++ni) {
      const int n = bn + wn * 32 + ni * 16 + lrow;
      const float bv = bias[n];
#pragma unroll
      for (int r = 0; r < 8; ++r) {
        const int m = bm + wm * 64 + mi * 16 + r + lhi * 8;
        const float v = (acc[mi][ni][r] + bv) * scale;
        if (OUT_MODE == 0) {
          ((float*)Cp)[(size_t)m * N + n] = v;
        } else {
          const int t = m >> 4, b = m & 15;   // m = t*B + b, B = 16
          const int h = n >> 6, d = n & 63;   // n = h*64 + d
          const size_t idx = (OUT_MODE == 1)
                                 ? ((size_t)((b * Hn + h) * Tn + t)) * Dn + d
                                 : ((size_t)((b * Hn + h) * Dn + d)) * Tn + t;
          ((_Float16*)Cp)[idx] = (_Float16)v;
        }
      }
    }
  }
}

// ---------------------------------------------------------------------------
// Flash-style attention. Grid = (B*H, T/128), block = 256 (8 waves).
// Wave w owns target rows [t0 + w*16, +16). Loops src in 128-chunks:
//   async-stage K tile (128x64 f16) and V tile (64x128 f16) into LDS (shared
//   by all 8 waves -> removes the 8x redundant global reads),
//   S = Q*K^T (16 WMMAs), + bias (streamed once from HBM), online softmax
//   (in-wave shfl_xor reductions), P staged to private LDS tile for the
//   C-layout -> A-layout transpose, then O += P*V (16 WMMAs, rescaled).
// q,k: [B,H,T,D] f16;  v: [B,H,D,T] f16;  out: [T,B,E] f16.
// LDS: 32KB P + 16KB K + 16KB V = 64KB (<< 320KB/WGP).
// ---------------------------------------------------------------------------
__global__ __launch_bounds__(256) void attn_fwd_kernel(
    const _Float16* __restrict__ qf, const _Float16* __restrict__ kf,
    const _Float16* __restrict__ vf, const float* __restrict__ bias,
    _Float16* __restrict__ attn_out) {
  __shared__ _Float16 kts[128 * 64];      // K tile  [s 0..127][d 0..63]
  __shared__ _Float16 vts[64 * 128];      // V tile  [d 0..63][s 0..127]
  __shared__ _Float16 pbuf[8 * 16 * 128]; // per-wave 16x128 P tile

  const int tid  = threadIdx.x;
  const int wave = tid >> 5;
  const int lane = tid & 31;
  const int lrow = lane & 15;
  const int lhi  = lane >> 4;
  const int bh   = blockIdx.x;  // b*H + h
  const int b    = bh / Hn;
  const int h    = bh % Hn;
  const int t0   = blockIdx.y * 128 + wave * 16;

  const _Float16* qb = qf + (size_t)bh * Tn * Dn;
  const _Float16* kb = kf + (size_t)bh * Tn * Dn;
  const _Float16* vb = vf + (size_t)bh * Dn * Tn;
  const float*    bb = bias + (size_t)bh * Tn * Tn;
  _Float16*       pl = pbuf + wave * (16 * 128);

  // Q fragments for this wave's 16 rows (K = D = 64 -> two 32-chunks)
  v16h qfr[2];
#pragma unroll
  for (int kc = 0; kc < 2; ++kc)
    qfr[kc] = load_frag16(qb + (size_t)(t0 + lrow) * Dn + kc * 32, lhi);

  v8f zero = {};
  float mrow[8], lsum[8];
  v8f oacc[4];
#pragma unroll
  for (int r = 0; r < 8; ++r) { mrow[r] = -3.0e38f; lsum[r] = 0.f; }
#pragma unroll
  for (int di = 0; di < 4; ++di) oacc[di] = zero;

  for (int s0 = 0; s0 < Tn; s0 += 128) {
    // ---- async-stage K and V tiles into LDS (ASYNCcnt) ----
    {
      const _Float16* kg = kb + (size_t)s0 * Dn;  // 128x64 tile, contiguous
#pragma unroll
      for (int i = 0; i < 4; ++i) {
        const int idx = (tid * 4 + i) * 8;        // half index, 16B chunks
        async_copy16(kg + idx, &kts[idx]);
      }
#pragma unroll
      for (int i = 0; i < 4; ++i) {
        const int c   = tid * 4 + i;              // chunk 0..1023
        const int row = c >> 4;                   // d row 0..63
        const int col = (c & 15) * 8;             // s col (halves)
        async_copy16(vb + (size_t)row * Tn + s0 + col, &vts[row * 128 + col]);
      }
      wait_async();       // own wave's async copies done
      __syncthreads();    // all waves' copies visible
    }

    // ---- S = Q * K^T over a 16(t) x 128(s) tile (K from LDS) ----
    v8f sacc[8];
#pragma unroll
    for (int ni = 0; ni < 8; ++ni) sacc[ni] = zero;
#pragma unroll
    for (int ni = 0; ni < 8; ++ni) {
#pragma unroll
      for (int kc = 0; kc < 2; ++kc) {
        v16h kfr = load_frag16(&kts[(ni * 16 + lrow) * Dn + kc * 32], lhi);
        sacc[ni] = __builtin_amdgcn_wmma_f32_16x16x32_f16(
            false, qfr[kc], false, kfr, (short)0, sacc[ni], false, false);
      }
    }

    // ---- bias add + online softmax (per C-layout row r) ----
#pragma unroll
    for (int r = 0; r < 8; ++r) {
      const int t = t0 + r + lhi * 8;
      float mx = -3.0e38f;
#pragma unroll
      for (int ni = 0; ni < 8; ++ni) {
        float v = sacc[ni][r] + bb[(size_t)t * Tn + s0 + ni * 16 + lrow];
        sacc[ni][r] = v;
        mx = fmaxf(mx, v);
      }
#pragma unroll
      for (int off = 1; off < 16; off <<= 1)
        mx = fmaxf(mx, __shfl_xor(mx, off, 32));
      const float nm    = fmaxf(mrow[r], mx);
      const float alpha = __expf(mrow[r] - nm);
      float sum = 0.f;
#pragma unroll
      for (int ni = 0; ni < 8; ++ni) {
        float p = __expf(sacc[ni][r] - nm);
        sacc[ni][r] = p;
        sum += p;
      }
#pragma unroll
      for (int off = 1; off < 16; off <<= 1)
        sum += __shfl_xor(sum, off, 32);
      lsum[r] = lsum[r] * alpha + sum;
      mrow[r] = nm;
#pragma unroll
      for (int di = 0; di < 4; ++di) oacc[di][r] *= alpha;
    }

    // ---- P tile to LDS (transpose C-layout -> A-layout) ----
    __syncthreads();  // WAR: K-tile reads + previous P reads complete
#pragma unroll
    for (int ni = 0; ni < 8; ++ni)
#pragma unroll
      for (int r = 0; r < 8; ++r)
        pl[(r + lhi * 8) * 128 + ni * 16 + lrow] = (_Float16)sacc[ni][r];
    __syncthreads();  // RAW: P writes visible before fragment reads

    // ---- O += P * V (V from LDS) ----
#pragma unroll
    for (int kc = 0; kc < 4; ++kc) {
      v16h pfr = load_frag16(pl + lrow * 128 + kc * 32, lhi);
#pragma unroll
      for (int di = 0; di < 4; ++di) {
        v16h vfr = load_frag16(&vts[(di * 16 + lrow) * 128 + kc * 32], lhi);
        oacc[di] = __builtin_amdgcn_wmma_f32_16x16x32_f16(
            false, pfr, false, vfr, (short)0, oacc[di], false, false);
      }
    }
    __syncthreads();  // V/P reads done before next chunk's async overwrites
  }

  // ---- normalize and store attn output as f16 [T,B,E] ----
#pragma unroll
  for (int di = 0; di < 4; ++di)
#pragma unroll
    for (int r = 0; r < 8; ++r) {
      const int t = t0 + r + lhi * 8;
      const int e = h * Dn + di * 16 + lrow;
      const float v = oacc[di][r] / lsum[r];
      attn_out[((size_t)t * Bn + b) * En + e] = (_Float16)v;
    }
}

// ---------------------------------------------------------------------------
extern "C" void kernel_launch(void* const* d_in, const int* in_sizes, int n_in,
                              void* d_out, int out_size, void* d_ws, size_t ws_size,
                              hipStream_t stream) {
  (void)in_sizes; (void)n_in; (void)out_size; (void)ws_size;
  const float* query     = (const float*)d_in[0];
  const float* attn_bias = (const float*)d_in[1];
  const float* q_w = (const float*)d_in[2];
  const float* q_b = (const float*)d_in[3];
  const float* k_w = (const float*)d_in[4];
  const float* k_b = (const float*)d_in[5];
  const float* v_w = (const float*)d_in[6];
  const float* v_b = (const float*)d_in[7];
  const float* out_w = (const float*)d_in[8];
  const float* out_b = (const float*)d_in[9];
  float* out = (float*)d_out;

  // Workspace: q [B,H,T,D], k [B,H,T,D], v [B,H,D,T], attn [T,B,E] — all f16.
  const size_t bhtd = (size_t)Bn * Hn * Tn * Dn;  // 6,291,456 elems each
  _Float16* qf = (_Float16*)d_ws;
  _Float16* kf = qf + bhtd;
  _Float16* vf = kf + bhtd;
  _Float16* at = vf + bhtd;

  dim3 blk(256);
  dim3 gproj(Mn / 128, En / 128);  // 64 x 6

  const float scaling = 0.125f;  // 1/sqrt(64)
  gemm_wmma_kernel<false, 1><<<gproj, blk, 0, stream>>>(query, q_w, q_b, qf, Mn, En, En, scaling);
  gemm_wmma_kernel<false, 1><<<gproj, blk, 0, stream>>>(query, k_w, k_b, kf, Mn, En, En, 1.0f);
  gemm_wmma_kernel<false, 2><<<gproj, blk, 0, stream>>>(query, v_w, v_b, vf, Mn, En, En, 1.0f);

  attn_fwd_kernel<<<dim3(Bn * Hn, Tn / 128), blk, 0, stream>>>(qf, kf, vf, attn_bias, at);

  gemm_wmma_kernel<true, 0><<<gproj, blk, 0, stream>>>(at, out_w, out_b, out, Mn, En, En, 1.0f);
}
